// AttnBlock_14671608283197
// MI455X (gfx1250) — matrix operand order repaired and checked by reading, not compiled, not run on hardware
//
#include <hip/hip_runtime.h>
#include <math.h>

// ---------------- problem constants (from reference) ----------------
#define B_ 4
#define C_ 256
#define N_ 4096          // 64*64
#define G_ 32            // groups
#define SCALE_ 0.0625f   // C^-0.5 = 1/16

// ---------------- WMMA helpers (CDNA5 bf16, wave32) ----------------
typedef __attribute__((ext_vector_type(16))) __bf16 v16bf;
typedef __attribute__((ext_vector_type(8)))  __bf16 v8bf;
typedef __attribute__((ext_vector_type(8)))  float  v8f;

#define WMMA_BF16(a, b, c) \
  __builtin_amdgcn_wmma_f32_16x16x32_bf16(false, (a), false, (b), (short)0, (c), false, false)

static __device__ __forceinline__ v16bf mkfrag(v8bf lo, v8bf hi) {
  return __builtin_shufflevector(lo, hi, 0,1,2,3,4,5,6,7,8,9,10,11,12,13,14,15);
}
static __device__ __forceinline__ v8bf ld8(const __bf16* p) { return *(const v8bf*)p; }
static __device__ __forceinline__ v8f zero8() {
  v8f z;
#pragma unroll
  for (int i = 0; i < 8; ++i) z[i] = 0.f;
  return z;
}

// CDNA5 async global->LDS copy (16B per lane) + async counter waits.
static __device__ __forceinline__ void async_cp16(unsigned int lds_off, const void* g) {
  asm volatile("global_load_async_to_lds_b128 %0, %1, off" :: "v"(lds_off), "v"(g) : "memory");
}
#define WAIT_ASYNC_16() asm volatile("s_wait_asynccnt 0x10" ::: "memory")
#define WAIT_ASYNC_0()  asm volatile("s_wait_asynccnt 0x0"  ::: "memory")

// =====================================================================
// Kernel 0: convert qkv_w / proj_w fp32 -> bf16 in workspace (one-time)
// =====================================================================
__global__ __launch_bounds__(256) void k_wconv(const float* __restrict__ wq,
                                               const float* __restrict__ wp,
                                               unsigned short* __restrict__ wqb_u,
                                               unsigned short* __restrict__ wpb_u) {
  __bf16* wqb = reinterpret_cast<__bf16*>(wqb_u);
  __bf16* wpb = reinterpret_cast<__bf16*>(wpb_u);
  int i = blockIdx.x * 256 + threadIdx.x;
  if (i < 3 * C_ * C_) wqb[i] = (__bf16)wq[i];
  if (i < C_ * C_)     wpb[i] = (__bf16)wp[i];
}

// =====================================================================
// Kernel 1: GroupNorm stats: mu, rsigma per (batch, group)
// =====================================================================
__global__ __launch_bounds__(256) void k_gnstats(const float* __restrict__ x,
                                                 float* __restrict__ stats) {
  __shared__ float rs[256], rq[256];
  int bg = blockIdx.x;
  const float* xp = x + (size_t)bg * 8 * N_;
  float s = 0.f, q = 0.f;
  for (int i = threadIdx.x; i < 8 * N_; i += 256) {
    float v = xp[i];
    s += v; q += v * v;
  }
  rs[threadIdx.x] = s; rq[threadIdx.x] = q;
  __syncthreads();
  for (int off = 128; off > 0; off >>= 1) {
    if ((int)threadIdx.x < off) {
      rs[threadIdx.x] += rs[threadIdx.x + off];
      rq[threadIdx.x] += rq[threadIdx.x + off];
    }
    __syncthreads();
  }
  if (threadIdx.x == 0) {
    float inv_n = 1.f / (8.f * N_);
    float mu  = rs[0] * inv_n;
    float var = rq[0] * inv_n - mu * mu;
    stats[bg * 2 + 0] = mu;
    stats[bg * 2 + 1] = rsqrtf(var + 1e-6f);
  }
}

// =====================================================================
// Kernel 2: fused GroupNorm-affine + QKV GEMM (bf16 operands, f32 acc)
// 2-deep software pipeline over output chunks; dual accumulator chains
// to avoid WMMA->WMMA D->C dependency stalls.
// =====================================================================
__global__ __launch_bounds__(256, 1) void k_qkv(const float* __restrict__ x,
                                                const float* __restrict__ gamma,
                                                const float* __restrict__ beta,
                                                const float* __restrict__ stats,
                                                const unsigned short* __restrict__ wqb_u,
                                                const float* __restrict__ qkvb,
                                                unsigned short* __restrict__ qT_u,
                                                unsigned short* __restrict__ kT_u,
                                                unsigned short* __restrict__ vm_u) {
  const __bf16* wqb = reinterpret_cast<const __bf16*>(wqb_u);
  __bf16* qT = reinterpret_cast<__bf16*>(qT_u);
  __bf16* kT = reinterpret_cast<__bf16*>(kT_u);
  __bf16* vm = reinterpret_cast<__bf16*>(vm_u);

  __shared__ __align__(16) __bf16 h_lds[128][136];

  int b  = blockIdx.x >> 5;
  int n0 = (blockIdx.x & 31) * 128;
  int t = threadIdx.x, w = t >> 5, l = t & 31, ln = l & 15, kh = l >> 4;

  v16bf aF[8];
  for (int ph = 0; ph < 2; ++ph) {
    int c0p = ph * 128;
    __syncthreads();
    for (int i = t; i < 128 * 128; i += 256) {
      int c = c0p + (i >> 7);
      int n = i & 127;
      float mu = stats[(b * G_ + (c >> 3)) * 2 + 0];
      float rsg = stats[(b * G_ + (c >> 3)) * 2 + 1];
      float ga = gamma[c] * rsg;
      float be = beta[c] - mu * ga;
      float v  = x[((size_t)(b * C_ + c)) * N_ + n0 + n];
      h_lds[n][c - c0p] = (__bf16)(v * ga + be);
    }
    __syncthreads();
    int row = w * 16 + ln;
#pragma unroll
    for (int fc = 0; fc < 4; ++fc) {
      int cl = fc * 32 + kh * 8;
      aF[ph * 4 + fc] = mkfrag(ld8(&h_lds[row][cl]), ld8(&h_lds[row][cl + 16]));
    }
  }

  auto loadB = [&](v16bf* dst, int oc) {
    const __bf16* wrow = wqb + (size_t)(oc * 16 + ln) * C_;
#pragma unroll
    for (int f = 0; f < 8; ++f)
      dst[f] = mkfrag(ld8(wrow + f * 32 + kh * 8),
                      ld8(wrow + f * 32 + 16 + kh * 8));
  };
  auto cstore = [&](const v16bf* bF, int oc) {
    int o0 = oc * 16;
    v8f d0 = zero8(), d1 = zero8();
#pragma unroll
    for (int f = 0; f < 8; f += 2) {
      d0 = WMMA_BF16(aF[f],     bF[f],     d0);
      d1 = WMMA_BF16(aF[f + 1], bF[f + 1], d1);
    }
    v8f d = d0 + d1;
    float bias = qkvb[o0 + ln];
    int nb = n0 + w * 16 + kh * 8;
    if (o0 < 512) {
      __bf16* dst = (o0 < 256) ? qT : kT;
      int oo = ((o0 < 256) ? o0 : o0 - 256) + ln;
#pragma unroll
      for (int r = 0; r < 8; ++r)
        dst[((size_t)(b * N_ + nb + r)) * C_ + oo] = (__bf16)(d[r] + bias);
    } else {
      v8bf pk;
#pragma unroll
      for (int r = 0; r < 8; ++r) pk[r] = (__bf16)(d[r] + bias);
      *(v8bf*)&vm[((size_t)(b * C_ + (o0 - 512) + ln)) * N_ + nb] = pk;
    }
  };

  v16bf bA[8], bB[8];
  loadB(bA, 0);
  for (int oc = 0; oc < 48; oc += 2) {
    loadB(bB, oc + 1);
    cstore(bA, oc);
    if (oc + 2 < 48) loadB(bA, oc + 2);
    cstore(bB, oc + 1);
  }
}

// =====================================================================
// Kernel 3: flash attention, async double-buffered K/V staging.
// 128 threads (4 waves), 64 queries/WG, grid = B*N/64 = 256.
// Dynamic LDS layout (bytes):
//   [0      , 33792)  K tiles: 2 bufs x (32 keys x 264 bf16)
//   [33792  , 74752)  V tiles: 2 bufs x (256 c  x 40  bf16)
//   [74752  , 79872)  P relayout: 4 waves x (16 x 40 bf16)
// =====================================================================
#define KT_BUF_B   16896
#define VV_BASE_B  33792
#define VV_BUF_B   20480
#define PL_BASE_B  74752
#define ATTN_LDS_B 79872

__global__ __launch_bounds__(128, 1) void k_attn(const unsigned short* __restrict__ qT_u,
                                                 const unsigned short* __restrict__ kT_u,
                                                 const unsigned short* __restrict__ vm_u,
                                                 unsigned short* __restrict__ aT_u) {
  const __bf16* qT = reinterpret_cast<const __bf16*>(qT_u);
  const __bf16* kTp = reinterpret_cast<const __bf16*>(kT_u);
  const __bf16* vmp = reinterpret_cast<const __bf16*>(vm_u);
  __bf16* aT = reinterpret_cast<__bf16*>(aT_u);

  extern __shared__ __align__(16) char dsm[];
  const unsigned int lds0 = (unsigned int)(size_t)dsm;   // LDS byte address
  __bf16* KT = (__bf16*)dsm;
  __bf16* VV = (__bf16*)(dsm + VV_BASE_B);
  __bf16* PL = (__bf16*)(dsm + PL_BASE_B);

  int b    = blockIdx.x >> 6;
  int q0wg = (blockIdx.x & 63) * 64;
  int t = threadIdx.x, w = t >> 5, l = t & 31, ln = l & 15, kh = l >> 4;
  int q0 = q0wg + w * 16;

  // resident Q fragments (16 queries x 256 channels)
  v16bf aQ[8];
  {
    const __bf16* qrow = qT + ((size_t)(b * N_ + q0 + ln)) * C_;
#pragma unroll
    for (int f = 0; f < 8; ++f)
      aQ[f] = mkfrag(ld8(qrow + f * 32 + kh * 8),
                     ld8(qrow + f * 32 + 16 + kh * 8));
  }

  float mrow[8], lrow[8];
  v8f acc[16];
#pragma unroll
  for (int r = 0; r < 8; ++r) { mrow[r] = -INFINITY; lrow[r] = 0.f; }
#pragma unroll
  for (int ct = 0; ct < 16; ++ct) acc[ct] = zero8();

  auto stage = [&](int buf, int k0) {
    for (int u = t; u < 1024; u += 128) {           // K: 32 keys x 256 c
      int row = u >> 5, cu = (u & 31) * 8;
      unsigned int loff = lds0 + (unsigned int)(buf * KT_BUF_B + (row * 264 + cu) * 2);
      async_cp16(loff, kTp + ((size_t)(b * N_ + k0 + row)) * C_ + cu);
    }
    for (int u = t; u < 1024; u += 128) {           // V: 256 c x 32 keys
      int c = u >> 2, cu = (u & 3) * 8;
      unsigned int loff = lds0 + (unsigned int)(VV_BASE_B + buf * VV_BUF_B + (c * 40 + cu) * 2);
      async_cp16(loff, vmp + ((size_t)(b * C_ + c)) * N_ + k0 + cu);
    }
  };

  stage(0, 0);
  for (int ci = 0; ci < N_ / 32; ++ci) {
    int k0 = ci * 32;
    int cur = ci & 1;
    if (ci + 1 < N_ / 32) {
      stage(cur ^ 1, k0 + 32);   // overlap next chunk's loads with compute
      WAIT_ASYNC_16();           // retire current chunk's 16 issues only
    } else {
      WAIT_ASYNC_0();
    }
    __syncthreads();

    const __bf16* ktb = KT + cur * (KT_BUF_B / 2);
    const __bf16* vvb = VV + cur * (VV_BUF_B / 2);

    // S = (Qt x K) * scale : preload all 16 K fragments, then 16 WMMAs
    // (two independent accumulator chains s0/s1)
    v16bf kb[16];
#pragma unroll
    for (int f = 0; f < 8; ++f) {
      const __bf16* r0 = ktb + ln * 264 + f * 32 + kh * 8;
      const __bf16* r1 = ktb + (16 + ln) * 264 + f * 32 + kh * 8;
      kb[f * 2]     = mkfrag(ld8(r0), ld8(r0 + 16));
      kb[f * 2 + 1] = mkfrag(ld8(r1), ld8(r1 + 16));
    }
    v8f s0 = zero8(), s1 = zero8();
#pragma unroll
    for (int f = 0; f < 8; ++f) {
      s0 = WMMA_BF16(aQ[f], kb[f * 2],     s0);
      s1 = WMMA_BF16(aQ[f], kb[f * 2 + 1], s1);
    }

    // online softmax over this chunk's 32 keys
#pragma unroll
    for (int r = 0; r < 8; ++r) {
      float a0 = s0[r] * SCALE_, a1 = s1[r] * SCALE_;
      float mx = fmaxf(a0, a1);
      mx = fmaxf(mx, __shfl_xor(mx, 1));
      mx = fmaxf(mx, __shfl_xor(mx, 2));
      mx = fmaxf(mx, __shfl_xor(mx, 4));
      mx = fmaxf(mx, __shfl_xor(mx, 8));
      float mnew  = fmaxf(mrow[r], mx);
      float alpha = __expf(mrow[r] - mnew);
      mrow[r] = mnew;
      float p0 = __expf(a0 - mnew), p1 = __expf(a1 - mnew);
      float ps = p0 + p1;
      ps += __shfl_xor(ps, 1);
      ps += __shfl_xor(ps, 2);
      ps += __shfl_xor(ps, 4);
      ps += __shfl_xor(ps, 8);
      lrow[r] = lrow[r] * alpha + ps;
#pragma unroll
      for (int ct = 0; ct < 16; ++ct) acc[ct][r] *= alpha;
      PL[w * 640 + (r + 8 * kh) * 40 + ln]      = (__bf16)p0;
      PL[w * 640 + (r + 8 * kh) * 40 + 16 + ln] = (__bf16)p1;
    }

    // O += P x V^T : two groups of 8 preloaded fragments, 16 independent accs
    {
      const __bf16* pr = PL + w * 640 + ln * 40;
      v16bf pf = mkfrag(ld8(pr + kh * 8), ld8(pr + 16 + kh * 8));
#pragma unroll
      for (int g8 = 0; g8 < 2; ++g8) {
        v16bf bv[8];
#pragma unroll
        for (int j = 0; j < 8; ++j) {
          const __bf16* vb = vvb + ((g8 * 8 + j) * 16 + ln) * 40 + kh * 8;
          bv[j] = mkfrag(ld8(vb), ld8(vb + 16));
        }
#pragma unroll
        for (int j = 0; j < 8; ++j)
          acc[g8 * 8 + j] = WMMA_BF16(pf, bv[j], acc[g8 * 8 + j]);
      }
    }
    __syncthreads();   // all waves done reading before buffer is re-staged
  }

  // epilogue: normalize rows, store aT[b][q][c] bf16
  float inv[8];
#pragma unroll
  for (int r = 0; r < 8; ++r) inv[r] = 1.f / lrow[r];
#pragma unroll
  for (int ct = 0; ct < 16; ++ct)
#pragma unroll
    for (int r = 0; r < 8; ++r)
      aT[((size_t)(b * N_ + q0 + r + 8 * kh)) * C_ + ct * 16 + ln] =
          (__bf16)(acc[ct][r] * inv[r]);
}

// =====================================================================
// Kernel 4: out = proj_w @ a + proj_b + x  (pipelined, dual acc chains)
// =====================================================================
__global__ __launch_bounds__(256, 1) void k_proj(const unsigned short* __restrict__ aT_u,
                                                 const unsigned short* __restrict__ wpb_u,
                                                 const float* __restrict__ pb,
                                                 const float* __restrict__ x,
                                                 float* __restrict__ out) {
  const __bf16* aT  = reinterpret_cast<const __bf16*>(aT_u);
  const __bf16* wpb = reinterpret_cast<const __bf16*>(wpb_u);

  int b  = blockIdx.x >> 5;
  int n0 = (blockIdx.x & 31) * 128;
  int t = threadIdx.x, w = t >> 5, l = t & 31, ln = l & 15, kh = l >> 4;
  int nr = n0 + w * 16;

  v16bf aF[8];
  const __bf16* arow = aT + ((size_t)(b * N_ + nr + ln)) * C_;
#pragma unroll
  for (int f = 0; f < 8; ++f)
    aF[f] = mkfrag(ld8(arow + f * 32 + kh * 8),
                   ld8(arow + f * 32 + 16 + kh * 8));

  auto loadB = [&](v16bf* dst, int oc) {
    const __bf16* wrow = wpb + (size_t)(oc * 16 + ln) * C_;
#pragma unroll
    for (int f = 0; f < 8; ++f)
      dst[f] = mkfrag(ld8(wrow + f * 32 + kh * 8),
                      ld8(wrow + f * 32 + 16 + kh * 8));
  };
  auto cstore = [&](const v16bf* bF, int oc) {
    int o0 = oc * 16;
    v8f d0 = zero8(), d1 = zero8();
#pragma unroll
    for (int f = 0; f < 8; f += 2) {
      d0 = WMMA_BF16(aF[f],     bF[f],     d0);
      d1 = WMMA_BF16(aF[f + 1], bF[f + 1], d1);
    }
    v8f d = d0 + d1;
    float bias = pb[o0 + ln];
    size_t base = ((size_t)(b * C_ + o0 + ln)) * N_ + nr + kh * 8;
    float4 x0 = *(const float4*)(x + base);
    float4 x1 = *(const float4*)(x + base + 4);
    float4 r0, r1;
    r0.x = d[0] + bias + x0.x;  r0.y = d[1] + bias + x0.y;
    r0.z = d[2] + bias + x0.z;  r0.w = d[3] + bias + x0.w;
    r1.x = d[4] + bias + x1.x;  r1.y = d[5] + bias + x1.y;
    r1.z = d[6] + bias + x1.z;  r1.w = d[7] + bias + x1.w;
    *(float4*)(out + base)     = r0;
    *(float4*)(out + base + 4) = r1;
  };

  v16bf bA[8], bB[8];
  loadB(bA, 0);
  for (int oc = 0; oc < 16; oc += 2) {
    loadB(bB, oc + 1);
    cstore(bA, oc);
    if (oc + 2 < 16) loadB(bA, oc + 2);
    cstore(bB, oc + 1);
  }
}

// =====================================================================
// Host launcher
// =====================================================================
extern "C" void kernel_launch(void* const* d_in, const int* in_sizes, int n_in,
                              void* d_out, int out_size, void* d_ws, size_t ws_size,
                              hipStream_t stream) {
  const float* x   = (const float*)d_in[0];
  const float* gsc = (const float*)d_in[1];
  const float* gbi = (const float*)d_in[2];
  const float* wq  = (const float*)d_in[3];
  const float* bq  = (const float*)d_in[4];
  const float* wp  = (const float*)d_in[5];
  const float* bp  = (const float*)d_in[6];
  float* out = (float*)d_out;

  char* ws = (char*)d_ws;
  size_t off = 0;
  float* stats = (float*)(ws + off);               off += (size_t)B_ * G_ * 2 * sizeof(float);
  off = (off + 255) & ~(size_t)255;
  unsigned short* wqb = (unsigned short*)(ws + off); off += (size_t)3 * C_ * C_ * 2;
  unsigned short* wpb = (unsigned short*)(ws + off); off += (size_t)C_ * C_ * 2;
  off = (off + 255) & ~(size_t)255;
  unsigned short* qT = (unsigned short*)(ws + off);  off += (size_t)B_ * N_ * C_ * 2;
  unsigned short* kT = (unsigned short*)(ws + off);  off += (size_t)B_ * N_ * C_ * 2;
  unsigned short* vm = (unsigned short*)(ws + off);  off += (size_t)B_ * N_ * C_ * 2;
  unsigned short* aT = (unsigned short*)(ws + off);  off += (size_t)B_ * N_ * C_ * 2;

  k_wconv  <<<768, 256, 0, stream>>>(wq, wp, wqb, wpb);
  k_gnstats<<<B_ * G_, 256, 0, stream>>>(x, stats);
  k_qkv    <<<B_ * (N_ / 128), 256, 0, stream>>>(x, gsc, gbi, stats, wqb, bq, qT, kT, vm);
  k_attn   <<<B_ * (N_ / 64), 128, ATTN_LDS_B, stream>>>(qT, kT, vm, aT);
  k_proj   <<<B_ * (N_ / 128), 256, 0, stream>>>(aT, wpb, bp, x, out);
}
